// SpanAttention_292057776139
// MI455X (gfx1250) — matrix operand order repaired and checked by reading, not compiled, not run on hardware
//
#include <hip/hip_runtime.h>
#include <hip/hip_bf16.h>
#include <cstdint>

typedef __attribute__((ext_vector_type(16))) _Float16 v16h;
typedef __attribute__((ext_vector_type(8)))  _Float16 v8h;
typedef __attribute__((ext_vector_type(8)))  float    v8f;
typedef __attribute__((ext_vector_type(4)))  unsigned int uint32x4;
typedef __attribute__((ext_vector_type(8)))  int          int32x8;
typedef __attribute__((ext_vector_type(4)))  int          int32x4;

#ifndef __has_builtin
#define __has_builtin(x) 0
#endif

#define H_    16
#define DH_   64
#define HID_  1024
#define B_    8
#define L_    8192
#define SPAN_ 64
#define NSP_  128          // L_/SPAN_
#define AS_   1032         // LDS A row stride (halves), padded
#define TS2_  136          // head-pair tile row stride (halves), padded

// ---- WMMA fragment loaders (CDNA5 16-bit layouts, ISA 7.12.2) ----

// A 16x32 f16: row = m0 + lane%16; lane<16: halves0-7=K0..7, halves8-15=K16..23
//              lane>=16: halves0-7=K8..15, halves8-15=K24..31
static __device__ __forceinline__ v16h ldsA_frag(const _Float16* Abase, int lda,
                                                 int m0, int k0, int lane) {
  int row   = m0 + (lane & 15);
  int khalf = (lane >> 4) << 3;  // 0 or 8
  const _Float16* p = Abase + (size_t)row * lda + k0 + khalf;
  union { v16h v; v8h h[2]; } u;
  u.h[0] = *(const v8h*)(p);
  u.h[1] = *(const v8h*)(p + 16);
  return u.v;
}

// B 32x16 f16 from transposed weights Bt[col][k]:
// lane%16 = N col; lanes0-15 hold K=kbase+0..7,16..23, lanes16-31 hold K=kbase+8..15,24..31
static __device__ __forceinline__ v16h glbB_frag(const _Float16* __restrict__ Bt,
                                                 int col, int kbase, int ldk, int lane) {
  int kb = kbase + ((lane >> 4) << 3);
  const _Float16* p = Bt + (size_t)col * ldk + kb;
  union { v16h v; v8h h[2]; } u;
  u.h[0] = *(const v8h*)(p);
  u.h[1] = *(const v8h*)(p + 16);
  return u.v;
}

static __device__ __forceinline__ v8f wmma_f16(v16h a, v16h b, v8f c) {
  return __builtin_amdgcn_wmma_f32_16x16x32_f16(false, a, false, b, (short)0, c, false, false);
}

// 64x1024(K) x 16-col strip GEMM with register double-buffering:
// next k-step's B + 4 A fragments are fetched BEFORE the current set's 4 WMMAs
// issue, so fragment registers are overwritten >=4 instructions after their last
// WMMA read (kills the F16-WMMA WAR v_nop padding, ISA 7.12.1).
static __device__ __forceinline__ void gemm_col_strip(const _Float16* Alds, int lda,
                                                      const _Float16* __restrict__ Bt,
                                                      int bcol, int ldk, int lane,
                                                      v8f acc[4]) {
  v16h b0 = glbB_frag(Bt, bcol, 0, ldk, lane);
  v16h a0[4], a1[4];
  #pragma unroll
  for (int mt = 0; mt < 4; ++mt) a0[mt] = ldsA_frag(Alds, lda, mt * 16, 0, lane);
  for (int kk = 0; kk < 32; kk += 2) {
    const int k1 = kk * 32 + 32;
    v16h b1 = glbB_frag(Bt, bcol, k1, ldk, lane);
    #pragma unroll
    for (int mt = 0; mt < 4; ++mt) a1[mt] = ldsA_frag(Alds, lda, mt * 16, k1, lane);
    #pragma unroll
    for (int mt = 0; mt < 4; ++mt) acc[mt] = wmma_f16(a0[mt], b0, acc[mt]);
    const int k2 = (kk * 32 + 64 < 1024) ? (kk * 32 + 64) : 0;  // clamped prefetch
    b0 = glbB_frag(Bt, bcol, k2, ldk, lane);
    #pragma unroll
    for (int mt = 0; mt < 4; ++mt) a0[mt] = ldsA_frag(Alds, lda, mt * 16, k2, lane);
    #pragma unroll
    for (int mt = 0; mt < 4; ++mt) acc[mt] = wmma_f16(a1[mt], b1, acc[mt]);
  }
}

// ---- Kernel 0: convert Wkv (1024x2048) and Wo (1024x1024) to transposed f16 ----
__global__ __launch_bounds__(256) void k_convert(const float* __restrict__ Wkv,
                                                 const float* __restrict__ Wo,
                                                 _Float16* __restrict__ WkvT,
                                                 _Float16* __restrict__ WoT) {
  const int NKV = 1024 * 2048;
  const int NWO = 1024 * 1024;
  for (int e = blockIdx.x * blockDim.x + threadIdx.x; e < NKV + NWO;
       e += gridDim.x * blockDim.x) {
    if (e < NKV) {
      int k = e >> 11, c = e & 2047;
      WkvT[(size_t)c * 1024 + k] = (_Float16)Wkv[e];
    } else {
      int e2 = e - NKV;
      int k = e2 >> 10, c = e2 & 1023;
      WoT[(size_t)c * 1024 + k] = (_Float16)Wo[e2];
    }
  }
}

// ---- Kernel 1: qh = q @ Wq + bq (8x1024, tiny) ----
__global__ __launch_bounds__(256) void k_qproj(const float* __restrict__ q,
                                               const float* __restrict__ Wq,
                                               const float* __restrict__ bq,
                                               float* __restrict__ qh) {
  int b = blockIdx.x;
  for (int j = 0; j < 4; ++j) {
    int c = threadIdx.x + j * 256;
    float s = bq[c];
    #pragma unroll 4
    for (int k = 0; k < HID_; ++k) s += q[b * HID_ + k] * Wq[k * HID_ + c];
    qh[b * HID_ + c] = s;
  }
}

// ---- Kernel 2: fused kv-projection + span attention, one WG per (b, span) ----
// Heads processed in PAIRS: 8 waves <-> 8 column tiles of 16; each wave owns the
// full 64-row strip (4 WMMAs per B fragment -> half the L2 B-traffic).
__global__ __launch_bounds__(256) void k_spanattn(const float* __restrict__ kv,
                                                  const unsigned char* __restrict__ mask,
                                                  const float* __restrict__ bkv,
                                                  const _Float16* __restrict__ WkvT,
                                                  const float* __restrict__ qh,
                                                  _Float16* __restrict__ midh) {
  __shared__ _Float16 Alds[64 * AS_];    // 132 KB: span rows (64) x K (1024) f16
  __shared__ _Float16 Tile2[64 * TS2_];  // 17 KB: head-pair 64x128 k or v tile (f16)
  __shared__ float    qbuf2[128];
  __shared__ float    sc2[128];
  __shared__ float    pbuf2[128];

  const int tid  = threadIdx.x;
  const int lane = tid & 31;
  const int w    = tid >> 5;             // wave id 0..7 == column tile
  const int lm   = lane & 15;
  const int mo   = (lane >> 4) << 3;     // C layout: M = r (+8 for upper lanes)
  const int b    = blockIdx.x >> 7;      // / NSP_
  const int n    = blockIdx.x & 127;

  // Stage kv[b, n*64 .. +63, :] into LDS as f16 (64 float4 per thread, coalesced)
  const float* src = kv + ((size_t)(b * L_ + n * SPAN_)) * HID_;
  for (int i = 0; i < 64; ++i) {
    int f   = tid + i * 256;             // float4 index, 16384 total
    int row = f >> 8;                    // 256 float4 per row
    int cf  = f & 255;
    float4 d = *(const float4*)(src + (size_t)row * HID_ + cf * 4);
    _Float16* dst = Alds + row * AS_ + cf * 4;
    dst[0] = (_Float16)d.x; dst[1] = (_Float16)d.y;
    dst[2] = (_Float16)d.z; dst[3] = (_Float16)d.w;
  }
  __syncthreads();

  const int midbase = (b * NSP_ + n) * (H_ * DH_);
  const int colT = w * 16 + lm;

  for (int hp = 0; hp < 8; ++hp) {       // head pair: heads 2hp, 2hp+1
    if (tid < 128) qbuf2[tid] = qh[b * HID_ + hp * 128 + tid] * 0.125f; // 1/sqrt(64)

    // ---- k pair: A(64x1024) @ Wkv[:, hp*128 .. +128) ----
    const int c0k = hp * 128 + w * 16;
    v8f kacc[4] = {};
    gemm_col_strip(Alds, AS_, WkvT, c0k + lm, HID_, lane, kacc);
    {
      float bias = bkv[c0k + lm];
      #pragma unroll
      for (int mt = 0; mt < 4; ++mt)
        #pragma unroll
        for (int r = 0; r < 8; ++r)
          Tile2[(mt * 16 + mo + r) * TS2_ + colT] = (_Float16)(kacc[mt][r] + bias);
    }
    __syncthreads();  // (A) k tile + qbuf visible

    // scores[hs][s] = sum_d k[s, hs*64+d] * qh_scaled[hs*64+d], masked
    if (tid < 128) {
      int hs = tid >> 6, s = tid & 63;
      float sum = 0.f;
      #pragma unroll 8
      for (int d = 0; d < 64; ++d)
        sum += (float)Tile2[s * TS2_ + hs * 64 + d] * qbuf2[hs * 64 + d];
      bool mk = mask[(size_t)b * L_ + n * SPAN_ + s] != 0;
      sc2[tid] = mk ? sum : -3.4028235e38f;
    }
    __syncthreads();  // (B) scores visible; Tile2 reads done

    // softmax per head over the 64-span (redundant reduction: trivial size)
    if (tid < 128) {
      int hs = tid >> 6;
      float mx = -3.4028235e38f;
      for (int i = 0; i < 64; ++i) mx = fmaxf(mx, sc2[hs * 64 + i]);
      float sm = 0.f;
      for (int i = 0; i < 64; ++i) sm += __expf(sc2[hs * 64 + i] - mx);
      pbuf2[tid] = __expf(sc2[tid] - mx) / sm;
    }

    // ---- v pair: A @ Wkv[:, 1024 + hp*128 .. +128) (overlaps softmax) ----
    const int c0v = HID_ + hp * 128 + w * 16;
    v8f vacc[4] = {};
    gemm_col_strip(Alds, AS_, WkvT, c0v + lm, HID_, lane, vacc);
    __syncthreads();  // (C) pbuf written; Tile2 free to overwrite
    {
      float bias = bkv[c0v + lm];
      #pragma unroll
      for (int mt = 0; mt < 4; ++mt)
        #pragma unroll
        for (int r = 0; r < 8; ++r)
          Tile2[(mt * 16 + mo + r) * TS2_ + colT] = (_Float16)(vacc[mt][r] + bias);
    }
    __syncthreads();  // (D) v tile visible

    if (tid < 128) {
      int hs = tid >> 6, d = tid & 63;
      float o = 0.f;
      #pragma unroll 8
      for (int s = 0; s < 64; ++s)
        o += pbuf2[hs * 64 + s] * (float)Tile2[s * TS2_ + hs * 64 + d];
      midh[midbase + hp * 128 + tid] = (_Float16)o;   // tid == hs*64+d
    }
    __syncthreads();  // (E) Tile2/pbuf/qbuf free for next head pair
  }
}

// ---- Kernel 3: out = mid(1024x1024 f16) @ Wo + bo ----
// A-chunk staging uses the Tensor Data Mover (2D tile, HW LDS padding to the
// 72-half stride the fragment reader expects), with a manual-copy fallback.
__global__ __launch_bounds__(256) void k_outproj(const _Float16* __restrict__ midh,
                                                 const _Float16* __restrict__ WoT,
                                                 const float* __restrict__ bo,
                                                 float* __restrict__ out) {
  __shared__ _Float16 As[64 * 72];       // 9 KB padded K-chunk of A (f16)
  const int tid = threadIdx.x, lane = tid & 31, w = tid >> 5, lm = lane & 15;
  const int R0 = blockIdx.x * 64, C0 = blockIdx.y * 64;
  const int mi = (w >> 2) * 2, ni = w & 3;
  const int m0a = mi * 16, m0b = m0a + 16;
  const int mo = (lane >> 4) << 3;

  v8f acc0 = {}; v8f acc1 = {};
  for (int kc = 0; kc < 16; ++kc) {
    __syncthreads();   // previous chunk's WMMAs done before As is overwritten
#if __has_builtin(__builtin_amdgcn_tensor_load_to_lds) && __has_builtin(__builtin_amdgcn_s_wait_tensorcnt)
    if (tid == 0) {
      // D# for a 64x64-half 2D tile, row stride 1024 halves, data_size=2B,
      // pad 4 DWORDs every 32 DWORDs -> LDS row stride 72 halves.
      unsigned long long ga =
          (unsigned long long)(uintptr_t)(midh + (size_t)R0 * HID_ + kc * 64);
      uint32x4 g0 = { 1u,                                   // count=1
                      0u,                                   // lds_addr = 0 (As @ 0)
                      (unsigned)(ga & 0xffffffffu),
                      (unsigned)((ga >> 32) & 0x01ffffffu) | 0x80000000u }; // type=2
      int32x8 g1 = { (int)((1u << 16) | (1u << 20) | (4u << 22) | (3u << 25)),
                     (int)(1024u << 16),                    // tensor_dim0 = 1024 (lo)
                     (int)(64u << 16),                      // dim0 hi=0 | tensor_dim1=64
                     (int)(64u << 16),                      // dim1 hi=0 | tile_dim0=64
                     64,                                    // tile_dim1=64 | tile_dim2=0
                     1024,                                  // tensor_dim0_stride (lo)
                     0, 0 };
      int32x4 g2 = { 0, 0, 0, 0 };
      int32x4 g3 = { 0, 0, 0, 0 };
      int32x8 g4 = { 0, 0, 0, 0, 0, 0, 0, 0 };             // unused trailing group
      __builtin_amdgcn_tensor_load_to_lds(g0, g1, g2, g3, g4, 0);
      __builtin_amdgcn_s_wait_tensorcnt(0);
    }
#else
    for (int i = 0; i < 2; ++i) {
      int f = tid + i * 256;             // v8h group id, 512 total
      int h0 = f * 8;
      int row = h0 >> 6, cf = h0 & 63;
      *(v8h*)(As + row * 72 + cf) =
          *(const v8h*)(midh + (size_t)(R0 + row) * HID_ + kc * 64 + cf);
    }
#endif
    __syncthreads();
    #pragma unroll
    for (int kk = 0; kk < 2; ++kk) {
      int k0 = kk * 32;
      v16h bf = glbB_frag(WoT, C0 + ni * 16 + lm, kc * 64 + k0, HID_, lane);
      v16h a0 = ldsA_frag(As, 72, m0a, k0, lane);
      v16h a1 = ldsA_frag(As, 72, m0b, k0, lane);
      acc0 = wmma_f16(a0, bf, acc0);
      acc1 = wmma_f16(a1, bf, acc1);
    }
  }
  int col = C0 + ni * 16 + lm;
  float bb = bo[col];
  #pragma unroll
  for (int r = 0; r < 8; ++r) {
    out[(size_t)(R0 + m0a + mo + r) * HID_ + col] = acc0[r] + bb;
    out[(size_t)(R0 + m0b + mo + r) * HID_ + col] = acc1[r] + bb;
  }
}

extern "C" void kernel_launch(void* const* d_in, const int* in_sizes, int n_in,
                              void* d_out, int out_size, void* d_ws, size_t ws_size,
                              hipStream_t stream) {
  (void)in_sizes; (void)n_in; (void)out_size; (void)ws_size;
  const float*         q    = (const float*)d_in[0];
  const float*         kv   = (const float*)d_in[1];
  const unsigned char* mask = (const unsigned char*)d_in[2];
  /* d_in[3] = span_size (== 64, compile-time constant here) */
  const float*         Wq   = (const float*)d_in[4];
  const float*         bq   = (const float*)d_in[5];
  const float*         Wkv  = (const float*)d_in[6];
  const float*         bkv  = (const float*)d_in[7];
  const float*         Wo   = (const float*)d_in[8];
  const float*         bo   = (const float*)d_in[9];
  float* out = (float*)d_out;

  // workspace: WkvT f16 (4MB) | WoT f16 (2MB) | qh f32 (32KB) | mid f16 (2MB)
  char* ws = (char*)d_ws;
  _Float16* WkvT = (_Float16*)ws;
  _Float16* WoT  = (_Float16*)(ws + (4u << 20));
  float*    qh   = (float*)(ws + (6u << 20));
  _Float16* midh = (_Float16*)(ws + (6u << 20) + (1u << 15));

  k_convert <<<2048, 256, 0, stream>>>(Wkv, Wo, WkvT, WoT);
  k_qproj   <<<B_, 256, 0, stream>>>(q, Wq, bq, qh);
  k_spanattn<<<B_ * NSP_, 256, 0, stream>>>(kv, mask, bkv, WkvT, qh, midh);
  k_outproj <<<dim3(16, 16), 256, 0, stream>>>(midh, WoT, bo, out);
}
